// SlotAttention_9079560864457
// MI455X (gfx1250) — matrix operand order repaired and checked by reading, not compiled, not run on hardware
//
#include <hip/hip_runtime.h>
#include <hip/hip_bf16.h>

typedef __attribute__((ext_vector_type(16))) __bf16 v16bf;
typedef __attribute__((ext_vector_type(8)))  __bf16 v8bf;
typedef __attribute__((ext_vector_type(8)))  float  v8f;

#define B_SZ    64
#define N_SZ    4096
#define F_SZ    64
#define D_SZ    64
#define KSLOT   8
#define ITERS   3

__device__ __forceinline__ float bfbits2f(unsigned int u16bits) {
  union { unsigned int i; float f; } c;
  c.i = u16bits << 16;
  return c.f;
}

__device__ __forceinline__ v16bf cat8(v8bf lo, v8bf hi) {
  return __builtin_shufflevector(lo, hi, 0,1,2,3,4,5,6,7,8,9,10,11,12,13,14,15);
}

__device__ __forceinline__ v8f wmma_bf16(v16bf a, v16bf b, v8f c) {
  return __builtin_amdgcn_wmma_f32_16x16x32_bf16(false, a, false, b, (short)0, c,
                                                 false, false);
}

// ---------------------------------------------------------------------------
// Kernel 1: fused LayerNorm(inputs) + K/V projection (bf16 WMMA), writes bf16
// k and v into workspace. grid = (N/128, B), block = 256 (8 waves, 16 rows ea)
// ---------------------------------------------------------------------------
__global__ __launch_bounds__(256) void kv_proj_kernel(
    const float* __restrict__ inputs, const float* __restrict__ Wk,
    const float* __restrict__ Wv, const float* __restrict__ ln_g,
    const float* __restrict__ ln_b, __bf16* __restrict__ kout,
    __bf16* __restrict__ vout)
{
  __shared__ __align__(64) __bf16 wk_s[D_SZ * F_SZ];
  __shared__ __align__(64) __bf16 wv_s[D_SZ * F_SZ];
  __shared__ __align__(64) __bf16 xs[8][16 * F_SZ];   // per-wave LN'd x tile
  __shared__ __align__(64) __bf16 cs[8][16 * D_SZ];   // per-wave C staging

  const int tid = threadIdx.x;
  for (int i = tid; i < D_SZ * F_SZ; i += 256) {
    wk_s[i] = (__bf16)Wk[i];
    wv_s[i] = (__bf16)Wv[i];
  }
  __syncthreads();

  const int wave = tid >> 5, lane = tid & 31;
  const int r = lane & 15, h = lane >> 4;     // 2 lanes per row
  const int batch = blockIdx.y;
  const int row0 = blockIdx.x * 128 + wave * 16;

  // LayerNorm over F=64, split across the two half-waves
  const float* xr = inputs + ((size_t)batch * N_SZ + row0 + r) * F_SZ + h * 32;
  float vals[32];
  float s = 0.f, s2 = 0.f;
  #pragma unroll
  for (int i = 0; i < 32; ++i) { float t = xr[i]; vals[i] = t; s += t; s2 += t * t; }
  s  += __shfl_xor(s, 16, 32);
  s2 += __shfl_xor(s2, 16, 32);
  const float mu   = s * (1.f / 64.f);
  const float rstd = rsqrtf(s2 * (1.f / 64.f) - mu * mu + 1e-5f);
  __bf16* xw = &xs[wave][r * F_SZ + h * 32];
  #pragma unroll
  for (int i = 0; i < 32; ++i)
    xw[i] = (__bf16)((vals[i] - mu) * rstd * ln_g[h * 32 + i] + ln_b[h * 32 + i]);
  __syncthreads();

  // A operand per ISA 16-bit 16x32 layout: row = lane&15, K chunks at
  // (lane>>4)*8 and +16 (first WMMA: f 0..31, second: f 32..63)
  const __bf16* arow = &xs[wave][r * F_SZ];
  const int o = h * 8;
  const v16bf a0 = cat8(*(const v8bf*)(arow + o),      *(const v8bf*)(arow + o + 16));
  const v16bf a1 = cat8(*(const v8bf*)(arow + o + 32), *(const v8bf*)(arow + o + 48));

  #pragma unroll
  for (int m = 0; m < 2; ++m) {
    const __bf16* W = m ? wv_s : wk_s;
    __bf16* outp = m ? vout : kout;
    #pragma unroll
    for (int ct = 0; ct < 4; ++ct) {              // 4 output-column tiles of 16
      // B operand: col = lane&15 -> W row (ct*16+col), K chunks at (lane>>4)*8 and +16
      const __bf16* wrow = &W[(ct * 16 + r) * F_SZ];
      v16bf b0 = cat8(*(const v8bf*)(wrow + h * 8),      *(const v8bf*)(wrow + h * 8 + 16));
      v16bf b1 = cat8(*(const v8bf*)(wrow + 32 + h * 8), *(const v8bf*)(wrow + 48 + h * 8));
      v8f c = {};
      c = wmma_bf16(a0, b0, c);
      c = wmma_bf16(a1, b1, c);
      // C layout: (row = i + 8*(lane>>4), col = lane&15)
      #pragma unroll
      for (int i = 0; i < 8; ++i)
        cs[wave][(i + 8 * h) * D_SZ + ct * 16 + r] = (__bf16)c[i];
    }
    // coalesced row-major store: 2 lanes per row, 32 bf16 (64B) each
    {
      const v16bf* src = (const v16bf*)&cs[wave][r * D_SZ + h * 32];
      __bf16* dst = outp + ((size_t)batch * N_SZ + row0 + r) * D_SZ + h * 32;
      v16bf t0 = src[0];
      v16bf t1 = src[1];
      *(v16bf*)dst       = t0;
      *((v16bf*)dst + 1) = t1;
    }
  }
}

// ---------------------------------------------------------------------------
// Kernel 2: full slot-attention iteration loop. One workgroup per batch,
// slots resident in LDS, single pass over N per iteration (colsum divides out)
// grid = B, block = 512 (16 waves)
// ---------------------------------------------------------------------------
__global__ __launch_bounds__(512) void slot_iter_kernel(
    const float* __restrict__ noise,  const float* __restrict__ mu_p,
    const float* __restrict__ logsig, const float* __restrict__ Wq,
    const float* __restrict__ w_ih,   const float* __restrict__ w_hh,
    const float* __restrict__ b_ih,   const float* __restrict__ b_hh,
    const float* __restrict__ w1,     const float* __restrict__ b1p,
    const float* __restrict__ w2,     const float* __restrict__ b2p,
    const float* __restrict__ lns_g,  const float* __restrict__ lns_b,
    const float* __restrict__ lnm_g,  const float* __restrict__ lnm_b,
    const __bf16* __restrict__ kbuf,  const __bf16* __restrict__ vbuf,
    float* __restrict__ out)
{
  __shared__ float slots[KSLOT][D_SZ];
  __shared__ float sn_s[KSLOT][D_SZ];
  __shared__ float stat[KSLOT][2];
  __shared__ __align__(64) __bf16 qs[16][D_SZ];   // rows 8..15 zero padded
  __shared__ float attn_s[16][16 * KSLOT];        // per-wave attn tile
  __shared__ float colsum[KSLOT];
  __shared__ float upd[KSLOT][D_SZ];
  __shared__ float hn[KSLOT][D_SZ];
  __shared__ float relu1[KSLOT][2 * D_SZ];

  const int tid  = threadIdx.x;
  const int wave = tid >> 5, lane = tid & 31;
  const int k = tid >> 6, d = tid & 63;           // 512 threads -> (k,d) in 8x64
  const int batch = blockIdx.x;

  const __bf16* kb = kbuf + (size_t)batch * N_SZ * D_SZ;
  const __bf16* vb = vbuf + (size_t)batch * N_SZ * D_SZ;

  // slots = mu + exp(log_sigma) * noise
  slots[k][d] = mu_p[d] +
      __expf(logsig[d]) * noise[((size_t)batch * KSLOT + k) * D_SZ + d];
  __syncthreads();

  for (int it = 0; it < ITERS; ++it) {
    // ---- LN(slots) -> q (scale 1/sqrt(D) folded into q) ----
    if (d == 0) {
      float s = 0.f, s2 = 0.f;
      for (int i = 0; i < D_SZ; ++i) { float t = slots[k][i]; s += t; s2 += t * t; }
      float m = s * (1.f / 64.f);
      stat[k][0] = m;
      stat[k][1] = rsqrtf(s2 * (1.f / 64.f) - m * m + 1e-5f);
    }
    __syncthreads();
    sn_s[k][d] = (slots[k][d] - stat[k][0]) * stat[k][1] * lns_g[d] + lns_b[d];
    qs[8 + k][d] = (__bf16)0.f;   // zero pad rows 8..15 of q
    __syncthreads();
    {
      float acc = 0.f;
      #pragma unroll 8
      for (int e = 0; e < D_SZ; ++e) acc += sn_s[k][e] * Wq[d * D_SZ + e];
      qs[k][d] = (__bf16)(acc * 0.125f);
    }
    upd[k][d] = 0.f;
    if (tid < KSLOT) colsum[tid] = 0.f;
    __syncthreads();

    // ---- one pass over N: logits (WMMA) -> softmax -> colsum + updates ----
    const int col = lane & 15, hh = lane >> 4;
    const __bf16* qrow = &qs[col][0];
    const v16bf bq0 = cat8(*(const v8bf*)(qrow + hh * 8),      *(const v8bf*)(qrow + hh * 8 + 16));
    const v16bf bq1 = cat8(*(const v8bf*)(qrow + 32 + hh * 8), *(const v8bf*)(qrow + 48 + hh * 8));

    float csl = 0.f;
    float ua[KSLOT][2];
    #pragma unroll
    for (int kk = 0; kk < KSLOT; ++kk) { ua[kk][0] = 0.f; ua[kk][1] = 0.f; }

    for (int t = wave; t < N_SZ / 16; t += 16) {
      const int n0 = t * 16;
      const __bf16* arow = kb + (size_t)(n0 + col) * D_SZ;
      const int o = hh * 8;
      v16bf a0 = cat8(*(const v8bf*)(arow + o),      *(const v8bf*)(arow + o + 16));
      v16bf a1 = cat8(*(const v8bf*)(arow + o + 32), *(const v8bf*)(arow + o + 48));
      v8f c = {};
      c = wmma_bf16(a0, bq0, c);
      c = wmma_bf16(a1, bq1, c);

      // softmax over slots: values live across the 16-lane group, mask k>=8
      float l[8], red[8];
      #pragma unroll
      for (int i = 0; i < 8; ++i) { l[i] = (col < 8) ? c[i] : -3.0e38f; red[i] = l[i]; }
      #pragma unroll
      for (int msk = 1; msk <= 8; msk <<= 1) {
        #pragma unroll
        for (int i = 0; i < 8; ++i)
          red[i] = fmaxf(red[i], __shfl_xor(red[i], msk, 32));
      }
      float ev[8], ssum[8];
      #pragma unroll
      for (int i = 0; i < 8; ++i) { ev[i] = __expf(l[i] - red[i]); ssum[i] = ev[i]; }
      #pragma unroll
      for (int msk = 1; msk <= 8; msk <<= 1) {
        #pragma unroll
        for (int i = 0; i < 8; ++i)
          ssum[i] += __shfl_xor(ssum[i], msk, 32);
      }
      if (col < KSLOT) {
        #pragma unroll
        for (int i = 0; i < 8; ++i) {
          float a = ev[i] / ssum[i] + 1e-8f;
          attn_s[wave][(i + 8 * hh) * KSLOT + col] = a;
          csl += a;
        }
      }
      // updates: all 32 lanes, each owns d = {2*lane, 2*lane+1}
      #pragma unroll 4
      for (int rr = 0; rr < 16; ++rr) {
        unsigned int pv = *(const unsigned int*)(vb + (size_t)(n0 + rr) * D_SZ + 2 * lane);
        float v0 = bfbits2f(pv & 0xffffu);
        float v1 = bfbits2f(pv >> 16);
        #pragma unroll
        for (int kk = 0; kk < KSLOT; ++kk) {
          float a = attn_s[wave][rr * KSLOT + kk];
          ua[kk][0] += a * v0;
          ua[kk][1] += a * v1;
        }
      }
    }
    if (col < KSLOT) atomicAdd(&colsum[col], csl);
    #pragma unroll
    for (int kk = 0; kk < KSLOT; ++kk) {
      atomicAdd(&upd[kk][2 * lane],     ua[kk][0]);
      atomicAdd(&upd[kk][2 * lane + 1], ua[kk][1]);
    }
    __syncthreads();

    // renormalize (colsum divides out of the linear update)
    upd[k][d] = upd[k][d] / colsum[k];
    __syncthreads();

    // ---- GRU cell, thread per (k,d) ----
    float gi[3], gh[3];
    #pragma unroll
    for (int g = 0; g < 3; ++g) {
      const int row = g * 64 + d;
      float ai = b_ih[row], ah = b_hh[row];
      #pragma unroll 8
      for (int e2 = 0; e2 < D_SZ; ++e2) {
        ai += upd[k][e2]   * w_ih[row * D_SZ + e2];
        ah += slots[k][e2] * w_hh[row * D_SZ + e2];
      }
      gi[g] = ai; gh[g] = ah;
    }
    float rg = 1.f / (1.f + __expf(-(gi[0] + gh[0])));
    float zg = 1.f / (1.f + __expf(-(gi[1] + gh[1])));
    float ng = tanhf(gi[2] + rg * gh[2]);
    float hnew = (1.f - zg) * ng + zg * slots[k][d];
    __syncthreads();
    slots[k][d] = hnew;
    __syncthreads();

    // ---- MLP with pre-LN and residual ----
    if (d == 0) {
      float s = 0.f, s2 = 0.f;
      for (int i = 0; i < D_SZ; ++i) { float t = slots[k][i]; s += t; s2 += t * t; }
      float m = s * (1.f / 64.f);
      stat[k][0] = m;
      stat[k][1] = rsqrtf(s2 * (1.f / 64.f) - m * m + 1e-5f);
    }
    __syncthreads();
    hn[k][d] = (slots[k][d] - stat[k][0]) * stat[k][1] * lnm_g[d] + lnm_b[d];
    __syncthreads();
    #pragma unroll
    for (int half = 0; half < 2; ++half) {
      const int j = half * 64 + d;
      float acc = b1p[j];
      #pragma unroll 8
      for (int e2 = 0; e2 < D_SZ; ++e2) acc += hn[k][e2] * w1[j * D_SZ + e2];
      relu1[k][j] = fmaxf(acc, 0.f);
    }
    __syncthreads();
    {
      float acc = b2p[d];
      #pragma unroll 8
      for (int e2 = 0; e2 < 2 * D_SZ; ++e2) acc += relu1[k][e2] * w2[d * 2 * D_SZ + e2];
      slots[k][d] += acc;
    }
    __syncthreads();
  }

  out[((size_t)batch * KSLOT + k) * D_SZ + d] = slots[k][d];
}

extern "C" void kernel_launch(void* const* d_in, const int* in_sizes, int n_in,
                              void* d_out, int out_size, void* d_ws, size_t ws_size,
                              hipStream_t stream) {
  const float* inputs = (const float*)d_in[0];
  const float* noise  = (const float*)d_in[1];
  const float* mu     = (const float*)d_in[2];
  const float* logsig = (const float*)d_in[3];
  const float* Wq     = (const float*)d_in[4];
  const float* Wk     = (const float*)d_in[5];
  const float* Wv     = (const float*)d_in[6];
  const float* gw_ih  = (const float*)d_in[7];
  const float* gw_hh  = (const float*)d_in[8];
  const float* gb_ih  = (const float*)d_in[9];
  const float* gb_hh  = (const float*)d_in[10];
  const float* w1     = (const float*)d_in[11];
  const float* b1     = (const float*)d_in[12];
  const float* w2     = (const float*)d_in[13];
  const float* b2     = (const float*)d_in[14];
  const float* lni_g  = (const float*)d_in[15];
  const float* lni_b  = (const float*)d_in[16];
  const float* lns_g  = (const float*)d_in[17];
  const float* lns_b  = (const float*)d_in[18];
  const float* lnm_g  = (const float*)d_in[19];
  const float* lnm_b  = (const float*)d_in[20];

  __bf16* kbuf = (__bf16*)d_ws;
  __bf16* vbuf = kbuf + (size_t)B_SZ * N_SZ * D_SZ;

  dim3 g1(N_SZ / 128, B_SZ);
  kv_proj_kernel<<<g1, 256, 0, stream>>>(inputs, Wk, Wv, lni_g, lni_b, kbuf, vbuf);

  slot_iter_kernel<<<B_SZ, 512, 0, stream>>>(
      noise, mu, logsig, Wq, gw_ih, gw_hh, gb_ih, gb_hh,
      w1, b1, w2, b2, lns_g, lns_b, lnm_g, lnm_b,
      kbuf, vbuf, (float*)d_out);
}
